// GIN_14671608283166
// MI455X (gfx1250) — matrix operand order repaired and checked by reading, not compiled, not run on hardware
//
#include <hip/hip_runtime.h>
#include <math.h>

typedef _Float16 hv16 __attribute__((ext_vector_type(16)));
typedef float    fv8  __attribute__((ext_vector_type(8)));
typedef unsigned int u32x4 __attribute__((ext_vector_type(4)));
typedef int          i32x8 __attribute__((ext_vector_type(8)));
typedef int          i32x4 __attribute__((ext_vector_type(4)));

#define GIN_NODES 100000
#define GIN_EDGES 3200000
#define GIN_F     128
#define GIN_G     512
#define GIN_C     10
#define GIN_EPS   1e-5f

// ---------------- prep: fold BN, transpose+convert weights to f16 ----------------
// W is [din=128][dout=128] row-major; Wt is [n=128][k=128] (column of W contiguous in k)
__global__ void gin_prep_wt(const float* __restrict__ W, _Float16* __restrict__ Wt) {
    int idx = blockIdx.x * blockDim.x + threadIdx.x;   // 0..16383
    int k = idx >> 7, n = idx & 127;
    Wt[n * 128 + k] = (_Float16)W[k * 128 + n];
}

// sc[0..127] = scale, sc[128..255] = bias  (BN eval-mode folded with linear bias)
__global__ void gin_prep_sc(const float* __restrict__ b, const float* __restrict__ g,
                            const float* __restrict__ be, const float* __restrict__ m,
                            const float* __restrict__ v, float* __restrict__ sc) {
    int j = threadIdx.x;                                // 128 threads
    float s = g[j] * rsqrtf(v[j] + GIN_EPS);
    sc[j] = s;
    sc[128 + j] = (b[j] - m[j]) * s + be[j];
}

__global__ void gin_zero(float* __restrict__ p, int n) {
    int i = blockIdx.x * blockDim.x + threadIdx.x;
    if (i < n) p[i] = 0.0f;
}

// ---------------- WMMA GEMM: Out[M x 128] = relu( (A (+A2)) @ W * s + c ) ----------------
// block = 256 threads = 8 waves; block covers 128 rows; K = N = 128.
__global__ __launch_bounds__(256) void gin_gemm128(
    const float* __restrict__ A, const float* __restrict__ A2,
    const _Float16* __restrict__ Wt,        // [n][k] f16, n-major, k contiguous
    const float* __restrict__ sc,           // [0:128) scale, [128:256) bias
    float* __restrict__ Out, int M)
{
    __shared__ _Float16 sW[128 * 128];      // 32 KB weight tile

#if __has_builtin(__builtin_amdgcn_tensor_load_to_lds)
    // ---- TDM: DMA the whole 128x128 f16 weight tile into LDS (wave 0 issues) ----
    if (threadIdx.x < 32) {
        unsigned lds  = (unsigned)(unsigned long long)(uintptr_t)sW; // low 32 bits = LDS offset
        unsigned long long ga = (unsigned long long)(uintptr_t)Wt;
        u32x4 g0;
        g0[0] = 1u;                               // count=1, is_restore=0, gather off
        g0[1] = lds;                              // lds_addr
        g0[2] = (unsigned)ga;                     // global_addr[31:0]
        g0[3] = (unsigned)((ga >> 32) & 0x01FFFFFFull) | (2u << 30); // addr[56:32] | type=2
        i32x8 g1;
        g1[0] = (int)(1u << 16);                  // workgroup_mask=0, data_size=1 (2 bytes)
        g1[1] = (int)((128u & 0xFFFFu) << 16);    // tensor_dim0[15:0] in bits 63:48
        g1[2] = (int)((128u & 0xFFFFu) << 16);    // tensor_dim0[31:16]=0 | tensor_dim1[15:0]
        g1[3] = (int)(128u << 16);                // tensor_dim1[31:16]=0 | tile_dim0=128
        g1[4] = (int)128u;                        // tile_dim1=128 | tile_dim2=0
        g1[5] = (int)128u;                        // tensor_dim0_stride[31:0] = 128
        g1[6] = 0;                                // stride0[47:32] | stride1[15:0]
        g1[7] = 0;                                // stride1[47:16]
        i32x4 gz4 = {0, 0, 0, 0};
        i32x8 gz8 = {0, 0, 0, 0, 0, 0, 0, 0};
        // 6-arg form (this toolchain): (g0, g1, g2, g3, extra, cpol)
        __builtin_amdgcn_tensor_load_to_lds(g0, g1, gz4, gz4, gz8, 0);
#if __has_builtin(__builtin_amdgcn_s_wait_tensorcnt)
        __builtin_amdgcn_s_wait_tensorcnt(0);
#else
        asm volatile("s_wait_tensorcnt 0x0" ::: "memory");
#endif
    }
#else
    {   // fallback: cooperative load of the weight tile
        const float4* srcp = (const float4*)Wt;
        float4* dstp = (float4*)sW;
        for (int i = threadIdx.x; i < (128 * 128 * 2) / 16; i += 256) dstp[i] = srcp[i];
    }
#endif
    __syncthreads();

    const int lane = threadIdx.x & 31;
    const int wave = threadIdx.x >> 5;
    const int g    = lane >> 4;             // lane-half selector
    const int lr   = lane & 15;             // row (A) / column (B,C,D) index
    const int rowBase = blockIdx.x * 128 + wave * 16;
    int rowA = rowBase + lr;
    if (rowA >= M) rowA = M - 1;            // clamp loads; stores are guarded

    fv8 acc[8] = {};                        // 8 N-tiles of 16x16, fp32 accumulate

    #pragma unroll
    for (int kc = 0; kc < 4; ++kc) {        // K chunks of 32
        // ---- A fragment (16-bit A 16x32 layout): elements 0..7 -> K = kc*32+8g+i,
        //      elements 8..15 -> K = kc*32+16+8g+(i-8)
        float av[16];
        const float* ap = A + (size_t)rowA * 128 + kc * 32 + 8 * g;
        *(float4*)(av + 0)  = *(const float4*)(ap + 0);
        *(float4*)(av + 4)  = *(const float4*)(ap + 4);
        *(float4*)(av + 8)  = *(const float4*)(ap + 16);
        *(float4*)(av + 12) = *(const float4*)(ap + 20);
        if (A2) {                           // fused h + agg for the GINConv layer
            float bv[16];
            const float* ap2 = A2 + (size_t)rowA * 128 + kc * 32 + 8 * g;
            *(float4*)(bv + 0)  = *(const float4*)(ap2 + 0);
            *(float4*)(bv + 4)  = *(const float4*)(ap2 + 4);
            *(float4*)(bv + 8)  = *(const float4*)(ap2 + 16);
            *(float4*)(bv + 12) = *(const float4*)(ap2 + 20);
            #pragma unroll
            for (int i = 0; i < 16; ++i) av[i] += bv[i];
        }
        hv16 afrag;
        #pragma unroll
        for (int i = 0; i < 16; ++i) afrag[i] = (_Float16)av[i];

        // ---- sweep the 8 column tiles.
        // B (32x16, 16-bit) layout: lane holds column lr; lane-half g holds K values
        // [kc*32 + 8g, +8) then [kc*32 + 16 + 8g, +8) -> two 16B reads from transposed W.
        #pragma unroll
        for (int nt = 0; nt < 8; ++nt) {
            hv16 bfrag;
            const _Float16* bp = &sW[(nt * 16 + lr) * 128 + kc * 32 + 8 * g];
            *(float4*)&bfrag[0] = *(const float4*)(bp + 0);
            *(float4*)&bfrag[8] = *(const float4*)(bp + 16);
            acc[nt] = __builtin_amdgcn_wmma_f32_16x16x32_f16(
                false, afrag, false, bfrag, (short)0, acc[nt], false, false);
        }
    }

    // ---- epilogue: folded BN scale/bias + ReLU.  D layout: elem e -> row e+8g, lane&15 -> col
    #pragma unroll
    for (int nt = 0; nt < 8; ++nt) {
        const int col = nt * 16 + lr;
        const float s = sc[col], c = sc[128 + col];
        #pragma unroll
        for (int e = 0; e < 8; ++e) {
            const int row = rowBase + e + 8 * g;
            if (row < M) {
                float o = acc[nt][e] * s + c;
                Out[(size_t)row * 128 + col] = o > 0.0f ? o : 0.0f;
            }
        }
    }
}

// ---------------- edge scatter-add: agg[dst] += h[src], 32 lanes/edge ----------------
__global__ void gin_scatter(const float* __restrict__ h, const int* __restrict__ src,
                            const int* __restrict__ dst, float* __restrict__ agg, int E)
{
    int t = blockIdx.x * blockDim.x + threadIdx.x;
    int e = t >> 5;
    if (e >= E) return;
    int q = (t & 31) * 4;
    int s = src[e], d = dst[e];
    float4 v = *(const float4*)(h + (size_t)s * 128 + q);
    float* o = agg + (size_t)d * 128 + q;
    __hip_atomic_fetch_add(o + 0, v.x, __ATOMIC_RELAXED, __HIP_MEMORY_SCOPE_AGENT);
    __hip_atomic_fetch_add(o + 1, v.y, __ATOMIC_RELAXED, __HIP_MEMORY_SCOPE_AGENT);
    __hip_atomic_fetch_add(o + 2, v.z, __ATOMIC_RELAXED, __HIP_MEMORY_SCOPE_AGENT);
    __hip_atomic_fetch_add(o + 3, v.w, __ATOMIC_RELAXED, __HIP_MEMORY_SCOPE_AGENT);
}

// ---------------- per-graph sum pooling (+ node count) ----------------
__global__ void gin_pool(const float* __restrict__ h, const int* __restrict__ batch,
                         float* __restrict__ P, float* __restrict__ cnt, int N)
{
    int t = blockIdx.x * blockDim.x + threadIdx.x;
    int n = t >> 7, f = t & 127;
    if (n >= N) return;
    int gid = batch[n];
    __hip_atomic_fetch_add(&P[gid * 128 + f], h[(size_t)n * 128 + f],
                           __ATOMIC_RELAXED, __HIP_MEMORY_SCOPE_AGENT);
    if (cnt != nullptr && f == 0)
        __hip_atomic_fetch_add(&cnt[gid], 1.0f, __ATOMIC_RELAXED, __HIP_MEMORY_SCOPE_AGENT);
}

// ---------------- final: out = log_softmax(P0@W0 + cnt*b0 + P1@W1 + b1) ----------------
__global__ void gin_final(const float* __restrict__ P0, const float* __restrict__ P1,
                          const float* __restrict__ cnt,
                          const float* __restrict__ W0, const float* __restrict__ b0,
                          const float* __restrict__ W1, const float* __restrict__ b1,
                          float* __restrict__ out)
{
    int gid = blockIdx.x;                 // 512 blocks x 32 threads
    int c = threadIdx.x;
    __shared__ float lg[16];
    if (c < GIN_C) {
        float acc = cnt[gid] * b0[c] + b1[c];
        for (int k = 0; k < 128; ++k) {
            acc += P0[gid * 128 + k] * W0[k * GIN_C + c]
                 + P1[gid * 128 + k] * W1[k * GIN_C + c];
        }
        lg[c] = acc;
    }
    __syncthreads();
    if (c < GIN_C) {
        float mx = lg[0];
        for (int i = 1; i < GIN_C; ++i) mx = fmaxf(mx, lg[i]);
        float se = 0.0f;
        for (int i = 0; i < GIN_C; ++i) se += __expf(lg[i] - mx);
        out[gid * GIN_C + c] = lg[c] - mx - __logf(se);
    }
}

// ---------------- host-side orchestration ----------------
extern "C" void kernel_launch(void* const* d_in, const int* in_sizes, int n_in,
                              void* d_out, int out_size, void* d_ws, size_t ws_size,
                              hipStream_t stream)
{
    (void)in_sizes; (void)n_in; (void)out_size; (void)ws_size;
    // setup_inputs() dict order (nested dicts in insertion order):
    const float* x     = (const float*)d_in[0];
    const int*   ei    = (const int*)  d_in[1];   // [2, E]: row0 = src, row1 = dst
    const int*   batch = (const int*)  d_in[2];
    // params.first_h: w1,b1,g1,be1,m1,v1,w2,b2,g2,be2,m2,v2
    const float* fh[12]; for (int i = 0; i < 12; ++i) fh[i] = (const float*)d_in[3 + i];
    const float* nn[12]; for (int i = 0; i < 12; ++i) nn[i] = (const float*)d_in[15 + i];
    const float* l0w = (const float*)d_in[27];
    const float* l0b = (const float*)d_in[28];
    const float* l1w = (const float*)d_in[29];
    const float* l1b = (const float*)d_in[30];
    float* out = (float*)d_out;

    // ---- workspace carve-up (256B aligned) ----
    char* ws = (char*)d_ws;
    size_t off = 0;
    auto carve = [&](size_t bytes) -> void* {
        void* p = ws + off; off += (bytes + 255) & ~(size_t)255; return p;
    };
    const size_t NF = (size_t)GIN_NODES * GIN_F * sizeof(float);     // 51.2 MB
    float*     tmp  = (float*)carve(NF);            // MLP intermediate
    float*     h    = (float*)carve(NF);            // first_h output
    float*     agg  = (float*)carve(NF);            // neighbor sum, later reused as h2
    _Float16*  wtF1 = (_Float16*)carve(128*128*2);
    _Float16*  wtF2 = (_Float16*)carve(128*128*2);
    _Float16*  wtN1 = (_Float16*)carve(128*128*2);
    _Float16*  wtN2 = (_Float16*)carve(128*128*2);
    float*     scF1 = (float*)carve(256*4);
    float*     scF2 = (float*)carve(256*4);
    float*     scN1 = (float*)carve(256*4);
    float*     scN2 = (float*)carve(256*4);
    float*     P0   = (float*)carve((size_t)GIN_G*128*4);
    float*     P1   = (float*)carve((size_t)GIN_G*128*4);   // contiguous after P0
    float*     cnt  = (float*)carve(GIN_G*4);               // contiguous after P1

    // ---- 1. fold BN + transpose/convert weights ----
    gin_prep_wt<<<64, 256, 0, stream>>>(fh[0], wtF1);
    gin_prep_wt<<<64, 256, 0, stream>>>(fh[6], wtF2);
    gin_prep_wt<<<64, 256, 0, stream>>>(nn[0], wtN1);
    gin_prep_wt<<<64, 256, 0, stream>>>(nn[6], wtN2);
    gin_prep_sc<<<1, 128, 0, stream>>>(fh[1], fh[2], fh[3], fh[4],  fh[5],  scF1);
    gin_prep_sc<<<1, 128, 0, stream>>>(fh[7], fh[8], fh[9], fh[10], fh[11], scF2);
    gin_prep_sc<<<1, 128, 0, stream>>>(nn[1], nn[2], nn[3], nn[4],  nn[5],  scN1);
    gin_prep_sc<<<1, 128, 0, stream>>>(nn[7], nn[8], nn[9], nn[10], nn[11], scN2);

    // ---- 2. zero atomic targets: agg, P0|P1|cnt (contiguous) ----
    gin_zero<<<(GIN_NODES * GIN_F + 255) / 256, 256, 0, stream>>>(agg, GIN_NODES * GIN_F);
    gin_zero<<<(GIN_G * 128 * 2 + GIN_G + 255) / 256, 256, 0, stream>>>(P0, GIN_G * 128 * 2 + GIN_G);

    const int gemmBlocks = (GIN_NODES + 127) / 128;   // 782

    // ---- 3. h = MLP_first(x) ----
    gin_gemm128<<<gemmBlocks, 256, 0, stream>>>(x,   nullptr, wtF1, scF1, tmp, GIN_NODES);
    gin_gemm128<<<gemmBlocks, 256, 0, stream>>>(tmp, nullptr, wtF2, scF2, h,   GIN_NODES);

    // ---- 4. pool h -> P0, node counts ----
    gin_pool<<<(GIN_NODES * GIN_F) / 256, 256, 0, stream>>>(h, batch, P0, cnt, GIN_NODES);

    // ---- 5. agg = scatter_add(h[src] -> dst) ----
    const int* src = ei;
    const int* dst = ei + GIN_EDGES;
    gin_scatter<<<(GIN_EDGES * 32) / 256, 256, 0, stream>>>(h, src, dst, agg, GIN_EDGES);

    // ---- 6. h2 = MLP_nn0(h + agg); h2 reuses agg buffer ----
    gin_gemm128<<<gemmBlocks, 256, 0, stream>>>(agg, h, wtN1, scN1, tmp, GIN_NODES);
    gin_gemm128<<<gemmBlocks, 256, 0, stream>>>(tmp, nullptr, wtN2, scN2, agg, GIN_NODES);

    // ---- 7. pool h2 -> P1 ----
    gin_pool<<<(GIN_NODES * GIN_F) / 256, 256, 0, stream>>>(agg, batch, P1, nullptr, GIN_NODES);

    // ---- 8. classifier heads + log_softmax ----
    gin_final<<<GIN_G, 32, 0, stream>>>(P0, P1, cnt, l0w, l0b, l1w, l1b, out);
}
